// RVQQuantizer_19361712570766
// MI455X (gfx1250) — matrix operand order repaired and checked by reading, not compiled, not run on hardware
//
#include <hip/hip_runtime.h>
#include <hip/hip_bf16.h>

// ---------------------------------------------------------------------------
// Residual VQ forward for MI455X (gfx1250, wave32).
// latent [8,4096,64] f32, codebooks [8,1024,64] f32.
// Outputs (flat): z_q [N*64], codes [N*8] (as float), q_loss [1],
// new_codebooks [8*1024*64].
// Distance GEMM via v_wmma_f32_16x16x32_bf16 (4 WMMAs per LDS B-fetch),
// lane-private argmin in the hot loop, single end-of-wave butterfly merge.
// ---------------------------------------------------------------------------

typedef __attribute__((ext_vector_type(16))) __bf16 v16bf;
typedef __attribute__((ext_vector_type(4)))  __bf16 v4bf;
typedef __attribute__((ext_vector_type(8)))  float  v8f;

#define N_ROWS 32768          // B*T
#define DIM    64
#define QSTAGE 8
#define KCODE  1024
#define DECAYF 0.99f
#define CHUNK  256            // codewords staged in LDS per pass (32KB bf16)
#define ROWS_PER_WAVE 32      // two 16-row M tiles per wave
#define ROWS_PER_BLOCK 256    // 8 waves * 32 rows
#define THREADS 256

// ---- init: residual = latent, z_q = 0, q_loss = 0 -------------------------
__global__ void rvq_init(const float* __restrict__ latent,
                         float* __restrict__ residual,
                         float* __restrict__ zq,
                         float* __restrict__ qloss) {
  int i = blockIdx.x * blockDim.x + threadIdx.x;
  if (i < N_ROWS * DIM) {
    float v = latent[i];
    residual[i] = v;
    zq[i] = 0.0f;
  }
  if (i == 0) *qloss = 0.0f;
}

// ---- zero per-stage EMA stats --------------------------------------------
__global__ void rvq_zero_stats(float* __restrict__ counts,
                               float* __restrict__ embed) {
  int i = blockIdx.x * blockDim.x + threadIdx.x;
  if (i < KCODE) counts[i] = 0.0f;
  if (i < KCODE * DIM) embed[i] = 0.0f;
}

// ---- one RVQ stage: distances (WMMA) + argmin + quantize + stats ---------
__launch_bounds__(THREADS)
__global__ void rvq_stage(const float* __restrict__ cb,    // [K][D] this stage
                          float* __restrict__ residual,    // [N][D]
                          float* __restrict__ zq,          // [N][D] accum
                          float* __restrict__ codes,       // [N][Q]
                          float* __restrict__ counts,      // [K]
                          float* __restrict__ embed,       // [K][D]
                          float* __restrict__ qloss,
                          int q) {
  __shared__ __align__(128) __bf16 cb_lds[CHUNK][DIM];       // 32 KB
  __shared__ float cn_lds[CHUNK];                            // ||c||^2
  __shared__ int   widx[THREADS / 32][ROWS_PER_WAVE];        // best idx per row

  const int tid  = threadIdx.x;
  const int wave = tid >> 5;
  const int lane = tid & 31;
  const int half = lane >> 4;      // 0: lanes 0-15, 1: lanes 16-31
  const int l16  = lane & 15;

  const int row0 = blockIdx.x * ROWS_PER_BLOCK + wave * ROWS_PER_WAVE;

  // A operands: two 16x32 bf16 M-tiles (rows row0..row0+31), contiguous K order (B matches):
  //   lanes 0-15  -> row=l16, K in {0..15}
  //   lanes 16-31 -> row=l16, K in {16..31}
  const int kbase = half * 8;
  v16bf a[2][2];   // [mtile][ktile]
#pragma unroll
  for (int m = 0; m < 2; ++m) {
    const int arow = row0 + m * 16 + l16;
#pragma unroll
    for (int t = 0; t < 2; ++t) {
#pragma unroll
      for (int e = 0; e < 16; ++e) {
        int kk = half * 16 + e;
        a[m][t][e] = (__bf16)residual[arow * DIM + t * 32 + kk];
      }
    }
  }

  // lane-private running argmin: this lane only sees columns n = g*16 + l16,
  // half selects which 8 rows (M = v + 8*half) these accumulators cover.
  float best_d[2][8];
  int   best_i[2][8];
#pragma unroll
  for (int m = 0; m < 2; ++m)
#pragma unroll
    for (int v = 0; v < 8; ++v) { best_d[m][v] = 3.4e38f; best_i[m][v] = 0; }

  for (int c0 = 0; c0 < KCODE; c0 += CHUNK) {
    if (c0 + CHUNK < KCODE)
      __builtin_prefetch(&cb[(c0 + CHUNK) * DIM + tid], 0, 1);  // global_prefetch_b8
    __syncthreads();
    // stage one chunk: fp32 -> bf16 into LDS (vectorized) + squared norms
    {
      const float4* src4 = (const float4*)&cb[(c0 + tid) * DIM];
      float s = 0.0f;
#pragma unroll
      for (int d4 = 0; d4 < DIM / 4; ++d4) {
        float4 p = src4[d4];
        s += p.x * p.x + p.y * p.y + p.z * p.z + p.w * p.w;
        v4bf pk = {(__bf16)p.x, (__bf16)p.y, (__bf16)p.z, (__bf16)p.w};
        *(v4bf*)&cb_lds[tid][d4 * 4] = pk;                      // ds_store_b64
      }
      cn_lds[tid] = s;
    }
    __syncthreads();

    for (int g = 0; g < CHUNK / 16; ++g) {
      const int ncol = g * 16 + l16;  // this lane's column codeword (local)
      // B operand: 32x16 bf16, lane holds column n=l16, K = half*16 + e.
      const v16bf b0 = *(const v16bf*)&cb_lds[ncol][0 * 32 + half * 16];
      const v16bf b1 = *(const v16bf*)&cb_lds[ncol][1 * 32 + half * 16];
      const float cn    = cn_lds[ncol];
      const int   nglob = c0 + ncol;
#pragma unroll
      for (int m = 0; m < 2; ++m) {
        v8f acc = {0.f, 0.f, 0.f, 0.f, 0.f, 0.f, 0.f, 0.f};
        acc = __builtin_amdgcn_wmma_f32_16x16x32_bf16(
            false, a[m][0], false, b0, (short)0, acc, false, false);
        acc = __builtin_amdgcn_wmma_f32_16x16x32_bf16(
            false, a[m][1], false, b1, (short)0, acc, false, false);
#pragma unroll
        for (int v = 0; v < 8; ++v) {
          // ||r||^2 dropped: constant per row, irrelevant to argmin.
          float nd = __builtin_fmaf(-2.0f, acc[v], cn);
          bool  t  = nd < best_d[m][v];
          best_d[m][v] = t ? nd : best_d[m][v];     // v_cndmask, branchless
          best_i[m][v] = t ? nglob : best_i[m][v];
        }
      }
    }
  }

  // single end-of-wave butterfly merge across the 16 lanes of each half
  // (xor masks 1,2,4,8 never cross the half boundary).
#pragma unroll
  for (int m = 0; m < 2; ++m)
#pragma unroll
    for (int v = 0; v < 8; ++v) {
      float d0 = best_d[m][v];
      int   i0 = best_i[m][v];
#pragma unroll
      for (int msk = 1; msk < 16; msk <<= 1) {
        float od = __shfl_xor(d0, msk, 32);
        int   oi = __shfl_xor(i0, msk, 32);
        bool  t  = od < d0;
        d0 = t ? od : d0;
        i0 = t ? oi : i0;
      }
      if (l16 == 0) widx[wave][m * 16 + half * 8 + v] = i0;
    }
  __syncthreads();

  // epilogue: quantize, update residual/zq/stats (rows are block-exclusive)
  float lsum = 0.0f;
  for (int t = lane; t < ROWS_PER_WAVE * DIM; t += 32) {
    int r = t >> 6, d = t & 63;
    int grow = row0 + r;
    int idx  = widx[wave][r];
    float cv   = cb[idx * DIM + d];
    float res  = residual[grow * DIM + d];
    float diff = res - cv;
    residual[grow * DIM + d] = diff;        // residual -= quant
    zq[grow * DIM + d] += cv;               // z_q_hard accumulation
    lsum += diff * diff;                    // commitment loss term
    atomicAdd(&embed[idx * DIM + d], res);  // segment_sum(residual)
    if (d == 0) {
      atomicAdd(&counts[idx], 1.0f);
      codes[grow * QSTAGE + q] = (float)idx;
    }
  }
#pragma unroll
  for (int m = 1; m < 32; m <<= 1) lsum += __shfl_xor(lsum, m, 32);
  if (lane == 0)
    atomicAdd(qloss, lsum * (1.0f / (float)(N_ROWS * DIM)));
}

// ---- EMA codebook update --------------------------------------------------
__global__ void rvq_finalize(const float* __restrict__ cb,
                             const float* __restrict__ counts,
                             const float* __restrict__ embed,
                             float* __restrict__ newcb) {
  int i = blockIdx.x * blockDim.x + threadIdx.x;
  if (i >= KCODE * DIM) return;
  int k = i >> 6;
  float cnt  = counts[k];
  float c    = cb[i];
  float mean = embed[i] / fmaxf(cnt, 1.0f);
  newcb[i] = (cnt > 0.0f) ? (DECAYF * c + (1.0f - DECAYF) * mean) : c;
}

extern "C" void kernel_launch(void* const* d_in, const int* in_sizes, int n_in,
                              void* d_out, int out_size, void* d_ws, size_t ws_size,
                              hipStream_t stream) {
  (void)in_sizes; (void)n_in; (void)out_size; (void)ws_size;
  const float* latent    = (const float*)d_in[0];   // [8,4096,64]
  const float* codebooks = (const float*)d_in[1];   // [8,1024,64]

  // d_out layout: z_q | codes | q_loss | new_codebooks
  float* zq    = (float*)d_out;                         // N*D
  float* codes = zq + (size_t)N_ROWS * DIM;             // N*Q
  float* qloss = codes + (size_t)N_ROWS * QSTAGE;       // 1
  float* newcb = qloss + 1;                             // Q*K*D

  // workspace: residual | counts | embed_sum  (~8.3 MB)
  float* residual = (float*)d_ws;
  float* counts   = residual + (size_t)N_ROWS * DIM;
  float* embed    = counts + KCODE;

  rvq_init<<<(N_ROWS * DIM + 255) / 256, 256, 0, stream>>>(latent, residual, zq, qloss);

  for (int q = 0; q < QSTAGE; ++q) {
    const float* cbq = codebooks + (size_t)q * KCODE * DIM;
    rvq_zero_stats<<<(KCODE * DIM + 255) / 256, 256, 0, stream>>>(counts, embed);
    rvq_stage<<<N_ROWS / ROWS_PER_BLOCK, THREADS, 0, stream>>>(
        cbq, residual, zq, codes, counts, embed, qloss, q);
    rvq_finalize<<<(KCODE * DIM + 255) / 256, 256, 0, stream>>>(
        cbq, counts, embed, newcb + (size_t)q * KCODE * DIM);
  }
}
